// SelfAttention_89945205113509
// MI455X (gfx1250) — matrix operand order repaired and checked by reading, not compiled, not run on hardware
//
#include <hip/hip_runtime.h>

#define EMBED 2048
#define HEADS 16
#define HD 128
#define BATCH 4
#define SEQ 2048

typedef __attribute__((ext_vector_type(16))) __bf16 v16bf;
typedef __attribute__((ext_vector_type(4))) __bf16 v4bf;
typedef __attribute__((ext_vector_type(8))) float v8f;

// ---- gfx1250 async global<->LDS path (guarded; falls back to VGPR copies) ----
#if defined(__HIP_DEVICE_COMPILE__) && __has_builtin(__builtin_amdgcn_global_load_async_to_lds_b128)
#define USE_ASYNC_LOAD 1
#else
#define USE_ASYNC_LOAD 0
#endif
#if defined(__HIP_DEVICE_COMPILE__) && __has_builtin(__builtin_amdgcn_global_store_async_from_lds_b128)
#define USE_ASYNC_STORE 1
#else
#define USE_ASYNC_STORE 0
#endif
#if defined(__HIP_DEVICE_COMPILE__) && __has_builtin(__builtin_amdgcn_s_wait_asynccnt)
#define ASYNC_WAIT() __builtin_amdgcn_s_wait_asynccnt(0)
#else
#define ASYNC_WAIT() ((void)0)
#endif

#if USE_ASYNC_LOAD || USE_ASYNC_STORE
typedef __attribute__((ext_vector_type(4))) int v4i;
typedef __attribute__((address_space(1))) v4i gv4i;  // global int4*
typedef __attribute__((address_space(3))) v4i lv4i;  // LDS int4*
#endif

// 16B global -> LDS (async DMA when available)
__device__ __forceinline__ void stage16(__bf16* dst_lds, const __bf16* src_glb) {
#if USE_ASYNC_LOAD
  __builtin_amdgcn_global_load_async_to_lds_b128((gv4i*)src_glb, (lv4i*)dst_lds, 0, 0);
#else
  *(uint4*)dst_lds = *(const uint4*)src_glb;
#endif
}

// 16B LDS -> global (async DMA when available)
__device__ __forceinline__ void unstage16(__bf16* dst_glb, const __bf16* src_lds) {
#if USE_ASYNC_STORE
  __builtin_amdgcn_global_store_async_from_lds_b128((gv4i*)dst_glb, (lv4i*)src_lds, 0, 0);
#else
  *(uint4*)dst_glb = *(const uint4*)src_lds;
#endif
}

__device__ __forceinline__ v8f wmma_bf16(v16bf a, v16bf b, v8f c) {
  return __builtin_amdgcn_wmma_f32_16x16x32_bf16(false, a, false, b, (short)0, c,
                                                 false, false);
}

__device__ __forceinline__ v8f zero8() {
  v8f z;
#pragma unroll
  for (int i = 0; i < 8; ++i) z[i] = 0.0f;
  return z;
}

// 8 fp32 -> 8 bf16 packed into one 16B store (v_cvt_pk_bf16_f32 x4 + b128 store).
union BF8 { __bf16 h[8]; uint4 u; };
__device__ __forceinline__ void cvt_store8(__bf16* dst, const float* src) {
  float4 a = *(const float4*)(src);
  float4 b = *(const float4*)(src + 4);
  BF8 t;
  t.h[0] = (__bf16)a.x; t.h[1] = (__bf16)a.y; t.h[2] = (__bf16)a.z; t.h[3] = (__bf16)a.w;
  t.h[4] = (__bf16)b.x; t.h[5] = (__bf16)b.y; t.h[6] = (__bf16)b.z; t.h[7] = (__bf16)b.w;
  *(uint4*)dst = t.u;
}

// A fragment: 16x32 bf16; lanes 0-15 -> M=lane, K {0..7,16..23}; lanes 16-31 -> K {8..15,24..31}.
__device__ __forceinline__ v16bf frag_a_bf(const __bf16* lds, int row0, int k0, int stride) {
  const int l = threadIdx.x & 31;
  const int m = l & 15;
  const int kb = (l >> 4) << 3;
  const __bf16* p = lds + (row0 + m) * stride + k0 + kb;
  v16bf a;
#pragma unroll
  for (int j = 0; j < 8; ++j) { a[j] = p[j]; a[j + 8] = p[j + 16]; }
  return a;
}

// B fragment (32x16), B[k][n] = lds[(n0+n)*stride + k0+k]; lanes 0-15 K {0..7,16..23}, 16-31 K {8..15,24..31}.
__device__ __forceinline__ v16bf frag_b_t(const __bf16* lds, int n0, int k0, int stride) {
  const int l = threadIdx.x & 31;
  const int n = l & 15;
  const int kb = (l >> 4) << 3;
  const __bf16* p = lds + (n0 + n) * stride + k0 + kb;
  v16bf b;
#pragma unroll
  for (int j = 0; j < 8; ++j) { b[j] = p[j]; b[j + 8] = p[j + 16]; }
  return b;
}

// ---------------------------------------------------------------------------
// Kernel 0: one-shot fp32 -> bf16 weight conversion (8 elems/thread).
// ---------------------------------------------------------------------------
__global__ __launch_bounds__(256) void cvt_kernel(const float* __restrict__ src,
                                                  __bf16* __restrict__ dst,
                                                  int ngroups) {
  const int i = blockIdx.x * 256 + threadIdx.x;
  if (i < ngroups) cvt_store8(dst + (size_t)i * 8, src + (size_t)i * 8);
}

// ---------------------------------------------------------------------------
// Kernel 1: per-head projection. Y[m, e] = sum_d X[m, d] * Wbf[e, d]
// Block: 64 rows x 128 cols, 4 waves, each wave = 16 rows x 8 col-tiles.
// ---------------------------------------------------------------------------
__global__ __launch_bounds__(128) void qkv_proj_kernel(
    const float* __restrict__ X, const __bf16* __restrict__ Wbf,
    __bf16* __restrict__ Y) {
  __shared__ __bf16 sW[HD * 136];   // reused as 64x136 output staging
  __shared__ __bf16 sA[64 * 40];
  const int wave = threadIdx.x >> 5;
  const int row0 = blockIdx.x * 64;

  for (int i = threadIdx.x; i < 128 * 16; i += 128) {
    int r = i >> 4, g = i & 15;
    stage16(&sW[r * 136 + g * 8], &Wbf[r * HD + g * 8]);
  }
  ASYNC_WAIT();

  v8f acc[8];
#pragma unroll
  for (int t = 0; t < 8; ++t) acc[t] = zero8();

  for (int kc = 0; kc < HD; kc += 32) {
    __syncthreads();
    for (int i = threadIdx.x; i < 64 * 4; i += 128) {
      int r = i >> 2, g = i & 3;
      cvt_store8(&sA[r * 40 + g * 8], &X[(size_t)(row0 + r) * HD + kc + g * 8]);
    }
    __syncthreads();
    v16bf a = frag_a_bf(sA, wave * 16, 0, 40);
#pragma unroll
    for (int t = 0; t < 8; ++t) {
      v16bf b = frag_b_t(sW, t * 16, kc, 136);
      acc[t] = wmma_bf16(a, b, acc[t]);
    }
  }

  __syncthreads();
  {
    const int l = threadIdx.x & 31;
    const int n = l & 15;
    const int rb = (l >> 4) << 3;
#pragma unroll
    for (int t = 0; t < 8; ++t)
#pragma unroll
      for (int r = 0; r < 8; ++r)
        sW[(wave * 16 + rb + r) * 136 + t * 16 + n] = (__bf16)acc[t][r];
  }
  __syncthreads();
  for (int i = threadIdx.x; i < 64 * 16; i += 128) {
    int r = i >> 4, g = i & 15;
    unstage16(&Y[(size_t)(row0 + r) * HD + g * 8], &sW[r * 136 + g * 8]);
  }
}

// ---------------------------------------------------------------------------
// Kernel 2: flash attention per (batch, head, 64-row q-tile). 4 waves.
// ---------------------------------------------------------------------------
__global__ __launch_bounds__(128) void attn_kernel(
    const __bf16* __restrict__ Qp, const __bf16* __restrict__ Kp,
    const __bf16* __restrict__ Vp, const int* __restrict__ mask,
    __bf16* __restrict__ Ctx) {
  __shared__ __bf16 sK[64 * 136];   // K chunk (also Q staging, also Ctx staging)
  __shared__ __bf16 sVt[HD * 72];   // V chunk transposed: [d][j]
  __shared__ float sS[64 * 68];     // scores
  __shared__ __bf16 sP[64 * 72];    // softmax probabilities (bf16)
  __shared__ float sM[64], sL[64], sAl[64];

  const int h = blockIdx.y;
  const int b = blockIdx.z;
  const int q0 = blockIdx.x * 64;
  const int wave = threadIdx.x >> 5;
  const float scale = 0.02209708691207961f;  // 1/sqrt(2048)
  const int rstride = HEADS * HD;

  // Stage Q tile through sK once (async DMA); keep A-fragments in registers.
  const size_t qbase = ((size_t)(b * SEQ + q0) * HEADS + h) * HD;
  for (int i = threadIdx.x; i < 64 * 16; i += 128) {
    int r = i >> 4, g = i & 15;
    stage16(&sK[r * 136 + g * 8], &Qp[qbase + (size_t)r * rstride + g * 8]);
  }
  if (threadIdx.x < 64) { sM[threadIdx.x] = -1e30f; sL[threadIdx.x] = 0.0f; }
  ASYNC_WAIT();
  __syncthreads();
  v16bf qf[4];
#pragma unroll
  for (int kk = 0; kk < 4; ++kk) qf[kk] = frag_a_bf(sK, wave * 16, kk * 32, 136);

  v8f acc[8];
#pragma unroll
  for (int t = 0; t < 8; ++t) acc[t] = zero8();

  for (int kc = 0; kc < SEQ; kc += 64) {
    __syncthreads();  // previous iteration's reads done before restaging
    const size_t kvbase = ((size_t)(b * SEQ + kc) * HEADS + h) * HD;
    for (int i = threadIdx.x; i < 64 * 16; i += 128) {
      int r = i >> 4, g = i & 15;
      stage16(&sK[r * 136 + g * 8], &Kp[kvbase + (size_t)r * rstride + g * 8]);
      BF8 t;
      t.u = *(const uint4*)&Vp[kvbase + (size_t)r * rstride + g * 8];
#pragma unroll
      for (int e = 0; e < 8; ++e) sVt[(g * 8 + e) * 72 + r] = t.h[e];
    }
    if (kc + 64 < SEQ) {  // prefetch next chunk (global_prefetch_b8)
      const size_t nb = ((size_t)(b * SEQ + kc + 64) * HEADS + h) * HD;
      const int r = threadIdx.x & 63;
      const int half = (threadIdx.x >> 6) & 1;
      __builtin_prefetch(&Kp[nb + (size_t)r * rstride + half * 64], 0, 3);
      __builtin_prefetch(&Vp[nb + (size_t)r * rstride + half * 64], 0, 3);
    }
    ASYNC_WAIT();
    __syncthreads();

    // Score strip: wave's 16 q-rows x 64 keys.
    v8f sacc[4];
#pragma unroll
    for (int t = 0; t < 4; ++t) sacc[t] = zero8();
#pragma unroll
    for (int kk = 0; kk < 4; ++kk) {
#pragma unroll
      for (int t = 0; t < 4; ++t) {
        v16bf bf = frag_b_t(sK, t * 16, kk * 32, 136);
        sacc[t] = wmma_bf16(qf[kk], bf, sacc[t]);
      }
    }
    {
      const int l = threadIdx.x & 31;
      const int n = l & 15;
      const int rb = (l >> 4) << 3;
#pragma unroll
      for (int t = 0; t < 4; ++t)
#pragma unroll
        for (int r = 0; r < 8; ++r)
          sS[(wave * 16 + rb + r) * 68 + t * 16 + n] = sacc[t][r];
    }
    __syncthreads();

    // Online softmax: one thread per q-row. Mask applied before /sqrt (as ref).
    if (threadIdx.x < 64) {
      const int row = threadIdx.x;
      float* srow = &sS[row * 68];
      const int4* mp = (const int4*)&mask[b * SEQ + kc];
      const float mold = sM[row];
      float mc = -1e30f;
      for (int c4 = 0; c4 < 16; ++c4) {
        float4 s = *(float4*)&srow[c4 * 4];
        const int4 mk = mp[c4];
        s.x = ((mk.x == 0) ? -1e20f : s.x) * scale;
        s.y = ((mk.y == 0) ? -1e20f : s.y) * scale;
        s.z = ((mk.z == 0) ? -1e20f : s.z) * scale;
        s.w = ((mk.w == 0) ? -1e20f : s.w) * scale;
        *(float4*)&srow[c4 * 4] = s;
        mc = fmaxf(mc, fmaxf(fmaxf(s.x, s.y), fmaxf(s.z, s.w)));
      }
      const float mnew = fmaxf(mold, mc);
      const float alpha = __expf(mold - mnew);
      float lsum = 0.0f;
      for (int c4 = 0; c4 < 16; ++c4) {
        float4 s = *(float4*)&srow[c4 * 4];
        float p0 = __expf(s.x - mnew), p1 = __expf(s.y - mnew);
        float p2 = __expf(s.z - mnew), p3 = __expf(s.w - mnew);
        lsum += (p0 + p1) + (p2 + p3);
        v4bf pk = {(__bf16)p0, (__bf16)p1, (__bf16)p2, (__bf16)p3};
        *(v4bf*)&sP[row * 72 + c4 * 4] = pk;
      }
      sM[row] = mnew;
      sL[row] = sL[row] * alpha + lsum;
      sAl[row] = alpha;
    }
    __syncthreads();

    // Rescale running O and accumulate O += P @ V  (V^T staged -> frag_b_t).
    {
      const int l = threadIdx.x & 31;
      const int rb = (l >> 4) << 3;
      float alf[8];
#pragma unroll
      for (int r = 0; r < 8; ++r) alf[r] = sAl[wave * 16 + rb + r];
#pragma unroll
      for (int t = 0; t < 8; ++t)
#pragma unroll
        for (int r = 0; r < 8; ++r) acc[t][r] *= alf[r];
    }
#pragma unroll
    for (int kk = 0; kk < 2; ++kk) {
      v16bf pa = frag_a_bf(sP, wave * 16, kk * 32, 72);
#pragma unroll
      for (int t = 0; t < 8; ++t) {
        v16bf bf = frag_b_t(sVt, t * 16, kk * 32, 72);
        acc[t] = wmma_bf16(pa, bf, acc[t]);
      }
    }
  }

  // Normalize, stage into sK, async b128 write-out.
  {
    const int l = threadIdx.x & 31;
    const int n = l & 15;
    const int rb = (l >> 4) << 3;
#pragma unroll
    for (int r = 0; r < 8; ++r) {
      const int row = wave * 16 + rb + r;
      const float inv = 1.0f / sL[row];
#pragma unroll
      for (int t = 0; t < 8; ++t)
        sK[row * 136 + t * 16 + n] = (__bf16)(acc[t][r] * inv);
    }
  }
  __syncthreads();
  for (int i = threadIdx.x; i < 64 * 16; i += 128) {
    int r = i >> 4, g = i & 15;
    unstage16(&Ctx[(size_t)(b * SEQ + q0 + r) * EMBED + h * HD + g * 8],
              &sK[r * 136 + g * 8]);
  }
}

// ---------------------------------------------------------------------------
// Kernel 3: Out[m, e] = sum_c Ctx[m, c] * Wob[e, c] + bo[e]   (Wob = bf16 Wo)
// Block 128x128, 8 waves, each wave = 16 rows x 8 col-tiles, K-step 64.
// ---------------------------------------------------------------------------
__global__ __launch_bounds__(256) void out_proj_kernel(
    const __bf16* __restrict__ Ctx, const __bf16* __restrict__ Wob,
    const float* __restrict__ bo, float* __restrict__ Out) {
  __shared__ __bf16 sA[128 * 72];
  __shared__ __bf16 sB[128 * 72];
  const int m0 = blockIdx.x * 128;
  const int n0 = blockIdx.y * 128;
  const int wave = threadIdx.x >> 5;

  v8f acc[8];
#pragma unroll
  for (int t = 0; t < 8; ++t) acc[t] = zero8();

  for (int kc = 0; kc < EMBED; kc += 64) {
    __syncthreads();
    for (int i = threadIdx.x; i < 128 * 8; i += 256) {
      int r = i >> 3, g = i & 7;
      stage16(&sA[r * 72 + g * 8], &Ctx[(size_t)(m0 + r) * EMBED + kc + g * 8]);
      stage16(&sB[r * 72 + g * 8], &Wob[(size_t)(n0 + r) * EMBED + kc + g * 8]);
    }
    ASYNC_WAIT();
    __syncthreads();
#pragma unroll
    for (int kk = 0; kk < 2; ++kk) {
      v16bf a = frag_a_bf(sA, wave * 16, kk * 32, 72);
#pragma unroll
      for (int t = 0; t < 8; ++t) {
        v16bf bf = frag_b_t(sB, t * 16, kk * 32, 72);
        acc[t] = wmma_bf16(a, bf, acc[t]);
      }
    }
  }

  const int l = threadIdx.x & 31;
  const int n = l & 15;
  const int rb = (l >> 4) << 3;
#pragma unroll
  for (int t = 0; t < 8; ++t) {
    const float bias = bo[n0 + t * 16 + n];
#pragma unroll
    for (int r = 0; r < 8; ++r)
      Out[(size_t)(m0 + wave * 16 + rb + r) * EMBED + n0 + t * 16 + n] =
          acc[t][r] + bias;
  }
}

// ---------------------------------------------------------------------------
extern "C" void kernel_launch(void* const* d_in, const int* in_sizes, int n_in,
                              void* d_out, int out_size, void* d_ws, size_t ws_size,
                              hipStream_t stream) {
  (void)in_sizes; (void)n_in; (void)out_size; (void)ws_size;
  const float* value  = (const float*)d_in[0];
  const float* key_in = (const float*)d_in[1];
  const float* query  = (const float*)d_in[2];
  const int*   mask   = (const int*)d_in[3];
  const float* Wv     = (const float*)d_in[4];
  const float* Wk     = (const float*)d_in[5];
  const float* Wq     = (const float*)d_in[6];
  const float* Wo     = (const float*)d_in[7];
  const float* bo     = (const float*)d_in[8];
  float* out = (float*)d_out;

  const size_t elems = (size_t)BATCH * SEQ * EMBED;  // 16,777,216
  char* ws = (char*)d_ws;
  __bf16* Qp  = (__bf16*)(ws);
  __bf16* Kp  = (__bf16*)(ws + elems * 2);
  __bf16* Vp  = (__bf16*)(ws + elems * 4);
  __bf16* Ctx = (__bf16*)(ws + elems * 6);
  __bf16* Wqb = (__bf16*)(ws + elems * 8);                    // 3 x 32KB
  __bf16* Wkb = Wqb + HD * HD;
  __bf16* Wvb = Wkb + HD * HD;
  __bf16* Wob = (__bf16*)(ws);  // overlays Qp (dead after attention)

  // Convert small per-head weights once (fp32 -> bf16).
  cvt_kernel<<<dim3((HD * HD / 8 + 255) / 256), 256, 0, stream>>>(Wq, Wqb, HD * HD / 8);
  cvt_kernel<<<dim3((HD * HD / 8 + 255) / 256), 256, 0, stream>>>(Wk, Wkb, HD * HD / 8);
  cvt_kernel<<<dim3((HD * HD / 8 + 255) / 256), 256, 0, stream>>>(Wv, Wvb, HD * HD / 8);

  const int Mrows = BATCH * SEQ * HEADS;  // 131072 rows of 128
  dim3 gProj(Mrows / 64);
  qkv_proj_kernel<<<gProj, 128, 0, stream>>>(query,  Wqb, Qp);
  qkv_proj_kernel<<<gProj, 128, 0, stream>>>(key_in, Wkb, Kp);
  qkv_proj_kernel<<<gProj, 128, 0, stream>>>(value,  Wvb, Vp);

  dim3 gAttn(SEQ / 64, HEADS, BATCH);
  attn_kernel<<<gAttn, 128, 0, stream>>>(Qp, Kp, Vp, mask, Ctx);

  // Convert Wo once into the now-dead Qp region.
  const int wog = EMBED * EMBED / 8;
  cvt_kernel<<<dim3((wog + 255) / 256), 256, 0, stream>>>(Wo, Wob, wog);

  dim3 gOut((BATCH * SEQ) / 128, EMBED / 128);
  out_proj_kernel<<<gOut, 256, 0, stream>>>(Ctx, Wob, bo, out);
}
